// Attention_51316269253128
// MI455X (gfx1250) — matrix operand order repaired and checked by reading, not compiled, not run on hardware
//
#include <hip/hip_runtime.h>
#include <hip/hip_bf16.h>

// ---------------------------------------------------------------- types
typedef __bf16 bf16;
typedef __attribute__((ext_vector_type(16))) __bf16 v16bf;
typedef __attribute__((ext_vector_type(8)))  __bf16 v8bf;
typedef __attribute__((ext_vector_type(8)))  float  v8f;
typedef unsigned int u32;
typedef __attribute__((ext_vector_type(4))) u32 u32x4;
typedef __attribute__((ext_vector_type(8))) u32 u32x8;

#define B_  4
#define S_  1024
#define H_  32
#define D_  128
#define HID_ 4096
#define NEG_BIG (-1e30f)

// float -> bf16 (round-to-nearest-even), bit level
__device__ __forceinline__ bf16 f2bf(float x) {
    unsigned u = __builtin_bit_cast(unsigned, x);
    unsigned r = (u + 0x7FFFu + ((u >> 16) & 1u)) >> 16;
    return __builtin_bit_cast(bf16, (unsigned short)r);
}
__device__ __forceinline__ unsigned short f2bf_bits(float x) {
    unsigned u = __builtin_bit_cast(unsigned, x);
    return (unsigned short)((u + 0x7FFFu + ((u >> 16) & 1u)) >> 16);
}

// generic pointer -> 32-bit LDS byte offset (for TDM descriptors)
__device__ __forceinline__ u32 lds_off_u32(const void* p) {
    return (u32)(unsigned long long)(__attribute__((address_space(3))) const unsigned short*)p;
}

// ---------------------------------------------------------------- TDM: 2-D tile load Global -> LDS
// Loads a tile_y x tile_x (rows x elems) tile of 2-byte elements whose row stride is
// stride_elems, from gptr (tile origin), into LDS at lds_off.  When pad=true, inserts
// 4 dwords of LDS padding after every 16 dwords stored (64B data + 16B pad -> 80B rows
// for tile_x==32, bank-conflict-friendly).
__device__ __forceinline__ void tdm_load_2d(u32 lds_off, const void* gptr,
                                            u32 tile_x, u32 tile_y,
                                            u32 stride_elems, u32 tdim0, u32 tdim1,
                                            bool pad) {
    unsigned long long ga = (unsigned long long)gptr;
    u32x4 g0;
    g0[0] = 1u;                                   // count=1 valid descriptor
    g0[1] = lds_off;                              // lds_addr
    g0[2] = (u32)ga;                              // global_addr[31:0]
    g0[3] = (u32)((ga >> 32) & 0x1FFFFFFu) | (2u << 30);  // global_addr[56:32] | type=2
    u32x8 g1;
    u32 w0 = (1u << 16);                          // workgroup_mask=0, data_size=1 (2B)
    if (pad) w0 |= (1u << 20) | (3u << 22) | (3u << 25);  // pad_enable, 16 dwords, +4 dwords
    g1[0] = w0;
    g1[1] = (tdim0 & 0xFFFFu) << 16;              // tensor_dim0[15:0]  (bits 79:48)
    g1[2] = (tdim0 >> 16) | ((tdim1 & 0xFFFFu) << 16);  // tensor_dim0 hi | tensor_dim1 lo
    g1[3] = (tdim1 >> 16) | (tile_x << 16);       // tensor_dim1 hi | tile_dim0
    g1[4] = tile_y;                               // tile_dim1, tile_dim2=0
    g1[5] = stride_elems;                         // tensor_dim0_stride[31:0]
    g1[6] = 0;                                    // stride hi, dim1_stride lo
    g1[7] = 0;
    asm volatile("tensor_load_to_lds %0, %1" :: "s"(g0), "s"(g1) : "memory");
}

// ---------------------------------------------------------------- f32 -> bf16 convert
__global__ void cvt_bf16(const float* __restrict__ in, bf16* __restrict__ out, long n) {
    long i = (long)blockIdx.x * blockDim.x + threadIdx.x;
    if (i < n) out[i] = f2bf(in[i]);
}

// ---------------------------------------------------------------- GEMM: C(MxN) = A(MxK) * B(NxK)^T + bias
// block = 256 threads (8 waves); block tile 128x128; K-step 32.
// A & B 128x32 tiles staged into LDS by the Tensor Data Mover, double buffered:
// TDM of step kk+32 overlaps WMMA of step kk.  Padded LDS pitch = 40 elems (80B).
#define LPITCH 40
__global__ __launch_bounds__(256) void gemm_bf16_nt(const bf16* __restrict__ A,
                                                    const bf16* __restrict__ Bm,
                                                    const float* __restrict__ bias,
                                                    float* __restrict__ C,
                                                    int M, int N, int K) {
    __shared__ unsigned short smem[2][2][128 * LPITCH];   // [buf][A=0/B=1][row*pitch]

    const int lane = threadIdx.x & 31;
    const int w    = threadIdx.x >> 5;
    const int m0   = blockIdx.y * 128;
    const int n0   = blockIdx.x * 128;
    const int khalf16 = (lane >> 4) << 4;
    const int klo     = (lane >> 4) << 3;
    const int rbase   = (lane >> 4) << 3;

    const bf16* Abase = A  + (size_t)m0 * K;   // tile origin row m0, col kk
    const bf16* Bbase = Bm + (size_t)n0 * K;

    u32 offA[2] = { lds_off_u32(&smem[0][0][0]), lds_off_u32(&smem[1][0][0]) };
    u32 offB[2] = { lds_off_u32(&smem[0][1][0]), lds_off_u32(&smem[1][1][0]) };

    v8f acc[8] = {};

    // prologue: DMA first tiles into buffer 0
    if (w == 0) {
        tdm_load_2d(offA[0], Abase, 32, 128, (u32)K, (u32)K, (u32)M, true);
        tdm_load_2d(offB[0], Bbase, 32, 128, (u32)K, (u32)K, (u32)N, true);
        __builtin_amdgcn_s_wait_tensorcnt(0);
    }
    __syncthreads();

    int cur = 0;
    for (int kk = 0; kk < K; kk += 32) {
        // DMA next K-step tiles into the other buffer (freed by last barrier)
        if (w == 0 && kk + 32 < K) {
            tdm_load_2d(offA[cur ^ 1], Abase + kk + 32, 32, 128, (u32)K, (u32)K, (u32)M, true);
            tdm_load_2d(offB[cur ^ 1], Bbase + kk + 32, 32, 128, (u32)K, (u32)K, (u32)N, true);
        }

        const unsigned short* Asub = &smem[cur][0][0];
        const unsigned short* Bsub = &smem[cur][1][0];

        // A fragment: wave's 16 rows, two 16B chunks per lane
        const unsigned short* ar = Asub + (w * 16 + (lane & 15)) * LPITCH;
        v8bf alo = *(const v8bf*)(const void*)(ar + klo);
        v8bf ahi = *(const v8bf*)(const void*)(ar + 16 + klo);
        v16bf a;
#pragma unroll
        for (int i = 0; i < 8; ++i) { a[i] = alo[i]; a[i + 8] = ahi[i]; }

#pragma unroll
        for (int nt = 0; nt < 8; ++nt) {
            const unsigned short* br = Bsub + (nt * 16 + (lane & 15)) * LPITCH + klo;
            v8bf b0 = *(const v8bf*)(const void*)(br);
            v8bf b1 = *(const v8bf*)(const void*)(br + 16);
            v16bf b;
#pragma unroll
            for (int i = 0; i < 8; ++i) { b[i] = b0[i]; b[i + 8] = b1[i]; }
            acc[nt] = __builtin_amdgcn_wmma_f32_16x16x32_bf16(
                false, a, false, b, (short)0, acc[nt], false, false);
        }

        if (w == 0) __builtin_amdgcn_s_wait_tensorcnt(0);
        __syncthreads();
        cur ^= 1;
    }

#pragma unroll
    for (int nt = 0; nt < 8; ++nt) {
        int n = n0 + nt * 16 + (lane & 15);
        float bv = bias ? bias[n] : 0.f;
#pragma unroll
        for (int r = 0; r < 8; ++r) {
            int m = m0 + w * 16 + rbase + r;
            C[(size_t)m * N + n] = acc[nt][r] + bv;
        }
    }
}

// ---------------------------------------------------------------- RoPE helpers
__device__ __forceinline__ void rope_cs(int pos, int i, float& c, float& s) {
    float f = __expf(-((float)i / 32.f) * 9.210340371976184f) * (float)pos; // ln(10000)
    c = __cosf(f); s = __sinf(f);
}

// Q: (B,S,HID) f32  ->  rope'd (B,H,S,D) bf16
__global__ void rope_q_kernel(const float* __restrict__ Qf, const int* __restrict__ pos,
                              bf16* __restrict__ Qbf) {
    long idx = (long)blockIdx.x * blockDim.x + threadIdx.x;
    if (idx >= (long)B_ * H_ * S_ * D_) return;
    int d = idx & (D_ - 1);
    long t = idx >> 7;
    int s = (int)(t & (S_ - 1)); t >>= 10;
    int h = (int)(t & (H_ - 1)); t >>= 5;
    int b = (int)t;
    const float* x = Qf + ((size_t)(b * S_ + s) * HID_) + h * D_;
    int p = pos[b * S_ + s];
    float val;
    if (d < 32)       { float c, sn; rope_cs(p, d, c, sn);       val = x[d] * c - x[d + 32] * sn; }
    else if (d < 64)  { float c, sn; rope_cs(p, d - 32, c, sn);  val = x[d] * c + x[d - 32] * sn; }
    else              { val = x[d]; }
    Qbf[((size_t)(b * H_ + h) * S_ + s) * D_ + d] = f2bf(val);
}

// KV: (B,S,2D) f32 -> K rope'd (B,S,D) bf16 ; V transposed (B,D,S) bf16
__global__ void rope_kv_kernel(const float* __restrict__ KVf, const int* __restrict__ pos,
                               bf16* __restrict__ Kbf, bf16* __restrict__ Vt) {
    long idx = (long)blockIdx.x * blockDim.x + threadIdx.x;
    if (idx >= (long)B_ * S_ * D_) return;
    int d = idx & (D_ - 1);
    long t = idx >> 7;
    int s = (int)(t & (S_ - 1));
    int b = (int)(t >> 10);
    const float* x = KVf + (size_t)(b * S_ + s) * (2 * D_);
    int p = pos[b * S_ + s];
    float kv;
    if (d < 32)       { float c, sn; rope_cs(p, d, c, sn);       kv = x[d] * c - x[d + 32] * sn; }
    else if (d < 64)  { float c, sn; rope_cs(p, d - 32, c, sn);  kv = x[d] * c + x[d - 32] * sn; }
    else              { kv = x[d]; }
    Kbf[(size_t)(b * S_ + s) * D_ + d] = f2bf(kv);
    Vt[((size_t)b * D_ + d) * S_ + s]  = f2bf(x[D_ + d]);
}

// ---------------------------------------------------------------- flash attention
// grid (S/128, H, B), block 256. Wave w handles 16 query rows; key tiles of 64.
#define PROW 72   // padded LDS row stride (bf16 elems)
__global__ __launch_bounds__(256) void flash_attn_kernel(const bf16* __restrict__ Qbf,
                                                         const bf16* __restrict__ Kbf,
                                                         const bf16* __restrict__ Vt,
                                                         bf16* __restrict__ Obf) {
    __shared__ unsigned short pbuf[8][16 * PROW];

    const int lane = threadIdx.x & 31;
    const int w    = threadIdx.x >> 5;
    const int b    = blockIdx.z;
    const int h    = blockIdx.y;
    const int q0   = blockIdx.x * 128 + w * 16;
    const int khalf16 = (lane >> 4) << 4;
    const int rbase   = (lane >> 4) << 3;
    const float scale = 0.08838834764831845f; // 1/sqrt(128)

    const bf16* Qh = Qbf + (size_t)(b * H_ + h) * S_ * D_;
    const bf16* Kb = Kbf + (size_t)b * S_ * D_;
    const bf16* Vb = Vt  + (size_t)b * D_ * S_;

    // Q fragments for the 16x128 strip: 4 K-chunks of 32
    const bf16* Qrow = Qh + (size_t)(q0 + (lane & 15)) * D_;
    int klo = (lane >> 4) << 3;
    v16bf aq[4];
#pragma unroll
    for (int c = 0; c < 4; ++c) {
        v8bf lo = *(const v8bf*)(Qrow + c * 32 + khalf16);
        v8bf hi = *(const v8bf*)(Qrow + c * 32 + 8 + khalf16);
#pragma unroll
        for (int i = 0; i < 8; ++i) { aq[c][i] = lo[i]; aq[c][i + 8] = hi[i]; }
    }

    v8f o[8] = {};
    float mrow[8], lrow[8];
#pragma unroll
    for (int r = 0; r < 8; ++r) { mrow[r] = NEG_BIG; lrow[r] = 0.f; }

    const int ntiles = 2 * (blockIdx.x + 1);   // keys < (blockIdx.x+1)*128
    for (int kt = 0; kt < ntiles; ++kt) {
        const int kb = kt * 64;

        // ---- scores: 16x64 via 16 WMMAs
        v8f sc[4] = {};
#pragma unroll
        for (int nt = 0; nt < 4; ++nt) {
            const bf16* Krow = Kb + (size_t)(kb + nt * 16 + (lane & 15)) * D_;
#pragma unroll
            for (int c = 0; c < 4; ++c) {
                v16bf bk = *(const v16bf*)(Krow + c * 32 + khalf16);
                sc[nt] = __builtin_amdgcn_wmma_f32_16x16x32_bf16(
                    false, aq[c], false, bk, (short)0, sc[nt], false, false);
            }
        }

        // ---- scale + causal mask + row max
        float tmax[8];
#pragma unroll
        for (int r = 0; r < 8; ++r) tmax[r] = NEG_BIG;
#pragma unroll
        for (int nt = 0; nt < 4; ++nt) {
            int key = kb + nt * 16 + (lane & 15);
#pragma unroll
            for (int r = 0; r < 8; ++r) {
                int qm = q0 + rbase + r;
                float v = (key <= qm) ? sc[nt][r] * scale : NEG_BIG;
                sc[nt][r] = v;
                tmax[r] = fmaxf(tmax[r], v);
            }
        }
#pragma unroll
        for (int r = 0; r < 8; ++r) {
            tmax[r] = fmaxf(tmax[r], __shfl_xor(tmax[r], 1, 32));
            tmax[r] = fmaxf(tmax[r], __shfl_xor(tmax[r], 2, 32));
            tmax[r] = fmaxf(tmax[r], __shfl_xor(tmax[r], 4, 32));
            tmax[r] = fmaxf(tmax[r], __shfl_xor(tmax[r], 8, 32));
        }

        // ---- online softmax update
        float alpha[8], psum[8];
#pragma unroll
        for (int r = 0; r < 8; ++r) {
            float mn = fmaxf(mrow[r], tmax[r]);
            alpha[r] = __expf(mrow[r] - mn);
            mrow[r] = mn;
            psum[r] = 0.f;
        }
#pragma unroll
        for (int nt = 0; nt < 4; ++nt)
#pragma unroll
            for (int r = 0; r < 8; ++r) {
                float p = __expf(sc[nt][r] - mrow[r]);
                sc[nt][r] = p;
                psum[r] += p;
            }
#pragma unroll
        for (int r = 0; r < 8; ++r) {
            psum[r] += __shfl_xor(psum[r], 1, 32);
            psum[r] += __shfl_xor(psum[r], 2, 32);
            psum[r] += __shfl_xor(psum[r], 4, 32);
            psum[r] += __shfl_xor(psum[r], 8, 32);
            lrow[r] = lrow[r] * alpha[r] + psum[r];
        }
#pragma unroll
        for (int dt = 0; dt < 8; ++dt)
#pragma unroll
            for (int r = 0; r < 8; ++r) o[dt][r] *= alpha[r];

        // ---- P acc-layout -> LDS -> A-frag layout (per-wave region)
#pragma unroll
        for (int nt = 0; nt < 4; ++nt)
#pragma unroll
            for (int r = 0; r < 8; ++r)
                pbuf[w][(rbase + r) * PROW + nt * 16 + (lane & 15)] = f2bf_bits(sc[nt][r]);
        __syncthreads();   // uniform: ntiles depends only on blockIdx.x

        // ---- O += P * V
        const unsigned short* prow = &pbuf[w][(lane & 15) * PROW];
#pragma unroll
        for (int c2 = 0; c2 < 2; ++c2) {
            v8bf lo = *(const v8bf*)(const void*)(prow + c2 * 32 + khalf16);
            v8bf hi = *(const v8bf*)(const void*)(prow + c2 * 32 + 8 + khalf16);
            v16bf ap;
#pragma unroll
            for (int i = 0; i < 8; ++i) { ap[i] = lo[i]; ap[i + 8] = hi[i]; }
#pragma unroll
            for (int dt = 0; dt < 8; ++dt) {
                v16bf bv = *(const v16bf*)(Vb + (size_t)(dt * 16 + (lane & 15)) * S_
                                           + kb + c2 * 32 + khalf16);
                o[dt] = __builtin_amdgcn_wmma_f32_16x16x32_bf16(
                    false, ap, false, bv, (short)0, o[dt], false, false);
            }
        }
        __syncthreads();
    }

    // ---- epilogue: O / l  -> bf16 (B,S,HID)
#pragma unroll
    for (int dt = 0; dt < 8; ++dt) {
#pragma unroll
        for (int r = 0; r < 8; ++r) {
            int qm = q0 + rbase + r;
            float val = o[dt][r] / lrow[r];
            Obf[(size_t)(b * S_ + qm) * HID_ + h * D_ + dt * 16 + (lane & 15)] = f2bf(val);
        }
    }
}

// ---------------------------------------------------------------- launch
extern "C" void kernel_launch(void* const* d_in, const int* in_sizes, int n_in,
                              void* d_out, int out_size, void* d_ws, size_t ws_size,
                              hipStream_t stream) {
    (void)in_sizes; (void)n_in; (void)out_size; (void)ws_size;

    const int*   pos  = (const int*)  d_in[0];
    const float* hid  = (const float*)d_in[1];
    const float* Wq   = (const float*)d_in[2];
    const float* bq   = (const float*)d_in[3];
    const float* Wkv  = (const float*)d_in[4];
    const float* bkv  = (const float*)d_in[5];
    const float* Wo   = (const float*)d_in[6];
    const float* bo   = (const float*)d_in[7];
    float* out        = (float*)d_out;

    const long nHid = (long)B_ * S_ * HID_;        // 16,777,216
    const long nWq  = (long)HID_ * HID_;           // 16,777,216
    const long nWkv = (long)2 * D_ * HID_;         // 1,048,576
    const long nKV  = (long)B_ * S_ * 2 * D_;      // 2,097,152/2
    const long nBSD = (long)B_ * S_ * D_;          // 524,288

    // workspace layout
    char* p = (char*)d_ws;
    bf16*  hid_bf = (bf16*)p;            p += nHid * 2;
    bf16*  wq_bf  = (bf16*)p;            p += nWq * 2;
    bf16*  wkv_bf = (bf16*)p;            p += nWkv * 2;
    bf16*  wo_bf  = (bf16*)p;            p += nWq * 2;
    float* qf     = (float*)p;           p += nHid * 4;
    float* kvf    = (float*)p;           p += nKV * 4;
    bf16*  qbf    = (bf16*)p;            p += nHid * 2;
    bf16*  kbf    = (bf16*)p;            p += nBSD * 2;
    bf16*  vt     = (bf16*)p;            p += nBSD * 2;
    bf16*  attn_bf= (bf16*)p;            p += nHid * 2;

    const int TB = 256;
    // 1) conversions
    cvt_bf16<<<dim3((nHid + TB - 1) / TB), dim3(TB), 0, stream>>>(hid, hid_bf, nHid);
    cvt_bf16<<<dim3((nWq  + TB - 1) / TB), dim3(TB), 0, stream>>>(Wq,  wq_bf,  nWq);
    cvt_bf16<<<dim3((nWkv + TB - 1) / TB), dim3(TB), 0, stream>>>(Wkv, wkv_bf, nWkv);
    cvt_bf16<<<dim3((nWq  + TB - 1) / TB), dim3(TB), 0, stream>>>(Wo,  wo_bf,  nWq);

    // 2) Q projection
    gemm_bf16_nt<<<dim3(HID_ / 128, (B_ * S_) / 128), dim3(TB), 0, stream>>>(
        hid_bf, wq_bf, bq, qf, B_ * S_, HID_, HID_);
    // 3) KV projection
    gemm_bf16_nt<<<dim3((2 * D_) / 128, (B_ * S_) / 128), dim3(TB), 0, stream>>>(
        hid_bf, wkv_bf, bkv, kvf, B_ * S_, 2 * D_, HID_);

    // 4) RoPE + repack
    rope_q_kernel<<<dim3((unsigned)(((long)B_ * H_ * S_ * D_) / TB)), dim3(TB), 0, stream>>>(
        qf, pos, qbf);
    rope_kv_kernel<<<dim3((unsigned)(nBSD / TB)), dim3(TB), 0, stream>>>(kvf, pos, kbf, vt);

    // 5) flash attention
    flash_attn_kernel<<<dim3(S_ / 128, H_, B_), dim3(TB), 0, stream>>>(qbf, kbf, vt, attn_bf);

    // 6) output projection -> d_out (fp32)
    gemm_bf16_nt<<<dim3(HID_ / 128, (B_ * S_) / 128), dim3(TB), 0, stream>>>(
        attn_bf, wo_bf, bo, out, B_ * S_, HID_, HID_);
}
